// TA_MODEL_77584289235168
// MI455X (gfx1250) — compile-verified, matrix operand order checked by reading, not run on hardware
//
#include <hip/hip_runtime.h>
#include <cmath>

typedef __bf16 bf16;
typedef __attribute__((ext_vector_type(16))) __bf16 v16bf;
typedef __attribute__((ext_vector_type(8)))  float  v8f;
typedef __attribute__((ext_vector_type(4)))  int    v4i;

constexpr int NB = 32, NL = 512, NIN = 64, NH = 256, ND = 512;

struct __align__(16) U4 { unsigned int a, b, c, d; };
union FragU { v16bf v; U4 q[2]; };

#if __has_builtin(__builtin_amdgcn_global_load_async_to_lds_b128) && __has_builtin(__builtin_amdgcn_s_wait_asynccnt)
#define HAVE_ASYNC_LDS 1
typedef __attribute__((address_space(1))) v4i* gv4p;   // global int4*
typedef __attribute__((address_space(3))) v4i* lv4p;   // LDS int4*
#else
#define HAVE_ASYNC_LDS 0
#endif

// Load one 16x32 (A) or 32x16 (B) bf16 WMMA fragment from row-major storage.
// lane&15 selects the row (A) / column (B); khalf = lane>>4 selects K-half.
// Per ISA layout: low lanes hold K = [0..7] and [16..23], high lanes [8..15], [24..31].
__device__ __forceinline__ v16bf ldfrag(const bf16* p, int khalf) {
  FragU f;
  f.q[0] = *(const U4*)(p + (khalf << 3));
  f.q[1] = *(const U4*)(p + 16 + (khalf << 3));
  return f.v;
}

__device__ __forceinline__ v8f wmma_bf16(v16bf a, v16bf b, v8f c) {
  return __builtin_amdgcn_wmma_f32_16x16x32_bf16(false, a, false, b, (short)0, c, false, false);
}

__device__ __forceinline__ v8f zero8() { v8f a; for (int i = 0; i < 8; ++i) a[i] = 0.f; return a; }
__device__ __forceinline__ float sigm(float x) { return 1.f / (1.f + __expf(-x)); }

// ---------------------------------------------------------------- cvt f32->bf16
__global__ void cvt_bf16_kernel(const float* s, bf16* d, int n) {
  int i = blockIdx.x * 256 + threadIdx.x;
  if (i < n) d[i] = (bf16)s[i];
}

// ---------------------------------------------------------------- LSTM (both dirs)
// grid = 4 blocks: bit1 = direction, bit0 = batch half. 512 threads = 16 waves.
// Whole recurrence in one kernel: h/c/gates in LDS, gate GEMM via WMMA,
// x_t tile staged via async global->LDS copy.
__global__ __launch_bounds__(512) void lstm_kernel(
    const bf16* __restrict__ x_bf, const int* __restrict__ lens,
    const float* __restrict__ bias_f, const float* __restrict__ bias_b,
    const bf16* __restrict__ wih_f, const bf16* __restrict__ whh_f,
    const bf16* __restrict__ wih_b, const bf16* __restrict__ whh_b,
    float* __restrict__ lstm_f32, bf16* __restrict__ lstm_bf)
{
  const int dir  = blockIdx.x >> 1;
  const int half = blockIdx.x & 1;
  const bf16*  wih  = dir ? wih_b  : wih_f;
  const bf16*  whh  = dir ? whh_b  : whh_f;
  const float* bias = dir ? bias_b : bias_f;

  __shared__ __align__(16) bf16 xt[16][64];       //  2 KB
  __shared__ __align__(16) bf16 hbf[16][256];     //  8 KB
  __shared__ float cst[16][256];                  // 16 KB
  __shared__ float gts[16][1024];                 // 64 KB

  const int tid  = threadIdx.x;
  const int lane = tid & 31, wave = tid >> 5;
  const int khalf = lane >> 4, ln = lane & 15;
  const int g0 = wave * 64 + ln;                  // first gate column this lane covers

  for (int u = tid; u < 16 * 256; u += 512) {
    int m = u >> 8, j = u & 255;
    hbf[m][j] = (bf16)0.0f;
    cst[m][j] = 0.0f;
  }
  __syncthreads();

  for (int t = 0; t < NL; ++t) {
    // ---- stage x_t (16 rows x 64 bf16 = 2KB) : 128 lanes x 16B, async copy to LDS
    if (tid < 128) {
      int m = tid >> 3, chunk = tid & 7;
      int b = half * 16 + m;
      int len = lens[b];
      bool v = (dir == 0) || (t < len);
      int tcur = dir ? (v ? (len - 1 - t) : 0) : t;
      const bf16* src = x_bf + ((size_t)b * NL + tcur) * NIN + chunk * 8;
      bf16* dst = &xt[m][chunk * 8];
      if (v) {
#if HAVE_ASYNC_LDS
        __builtin_amdgcn_global_load_async_to_lds_b128((gv4p)(void*)src, (lv4p)(void*)dst, 0, 0);
#else
        *(U4*)dst = *(const U4*)src;
#endif
      } else {
        U4 z; z.a = z.b = z.c = z.d = 0u;
        *(U4*)dst = z;
      }
    }
#if HAVE_ASYNC_LDS
    __builtin_amdgcn_s_wait_asynccnt(0);
#endif
    __syncthreads();

    // ---- gates = X_t @ Wih^T + H @ Whh^T + bias  (16 x 1024 via WMMA)
    // each wave owns a 16x64 strip: 4 accumulators, A-fragment shared.
    v8f acc[4];
    for (int nn = 0; nn < 4; ++nn) acc[nn] = zero8();
#pragma unroll 1
    for (int kk = 0; kk < NIN; kk += 32) {
      v16bf af = ldfrag(&xt[ln][0] + kk, khalf);
#pragma unroll
      for (int nn = 0; nn < 4; ++nn) {
        v16bf bf = ldfrag(wih + (size_t)(g0 + nn * 16) * NIN + kk, khalf);
        acc[nn] = wmma_bf16(af, bf, acc[nn]);
      }
    }
#pragma unroll 1
    for (int kk = 0; kk < NH; kk += 32) {
      v16bf af = ldfrag(&hbf[ln][0] + kk, khalf);
#pragma unroll
      for (int nn = 0; nn < 4; ++nn) {
        v16bf bf = ldfrag(whh + (size_t)(g0 + nn * 16) * NH + kk, khalf);
        acc[nn] = wmma_bf16(af, bf, acc[nn]);
      }
    }
#pragma unroll
    for (int nn = 0; nn < 4; ++nn) {
      int gcol = g0 + nn * 16;
      float bv = bias[gcol];
#pragma unroll
      for (int r = 0; r < 8; ++r) gts[r + khalf * 8][gcol] = acc[nn][r] + bv;
    }
    __syncthreads();

    // ---- cell update
    for (int u = tid; u < 16 * 256; u += 512) {
      int m = u >> 8, j = u & 255;
      int b = half * 16 + m;
      int len = lens[b];
      bool v = (t < len);
      float gi = gts[m][j], gf = gts[m][256 + j], gg = gts[m][512 + j], go = gts[m][768 + j];
      float cn = sigm(gf) * cst[m][j] + sigm(gi) * tanhf(gg);
      float hn = sigm(go) * tanhf(cn);
      if (v) {
        cst[m][j] = cn;
        hbf[m][j] = (bf16)hn;
        size_t tout = dir ? (size_t)(len - 1 - t) : (size_t)t;
        size_t idx = ((size_t)b * NL + tout) * ND + (size_t)dir * NH + j;
        lstm_f32[idx] = hn;
        lstm_bf[idx]  = (bf16)hn;
      }
    }
    __syncthreads();
  }
}

// ---------------------------------------------------------------- dm softmax
__global__ __launch_bounds__(256) void dm_kernel(
    const float* __restrict__ timed, const int* __restrict__ lens,
    const float* __restrict__ pa, const float* __restrict__ pb,
    bf16* __restrict__ dmout)
{
  int row = blockIdx.x;            // 0 .. B*L-1
  int b = row / NL, l = row % NL;
  int len = lens[b];
  float tl  = timed[(size_t)b * NL + l];
  float aab = fabsf(pa[0]);
  float bc  = pb[0];
  const float param = 1.0f / logf(24.0f + 2.7f);
  bool rowv = (l < len);

  __shared__ float red[256];
  float vals[2];
  for (int it = 0; it < 2; ++it) {
    int m = threadIdx.x + it * 256;
    float tm = timed[(size_t)b * NL + m];
    float dmv = 1.0f / logf(fabsf(tl - tm) + 2.7f);
    float maskt = (dmv >= param) ? 1.0f : 0.0f;
    float vm = (rowv && (m < len)) ? 1.0f : 0.0f;
    float z = dmv * maskt * vm;
    float s = 1.0f / (1.0f + __expf(-(aab * z + bc)));
    float v = maskt * s;
    vals[it] = (m < len) ? v : -1e9f;
  }
  float mx = fmaxf(vals[0], vals[1]);
  red[threadIdx.x] = mx; __syncthreads();
  for (int s = 128; s > 0; s >>= 1) {
    if (threadIdx.x < s) red[threadIdx.x] = fmaxf(red[threadIdx.x], red[threadIdx.x + s]);
    __syncthreads();
  }
  mx = red[0]; __syncthreads();
  float e0 = __expf(vals[0] - mx), e1 = __expf(vals[1] - mx);
  red[threadIdx.x] = e0 + e1; __syncthreads();
  for (int s = 128; s > 0; s >>= 1) {
    if (threadIdx.x < s) red[threadIdx.x] += red[threadIdx.x + s];
    __syncthreads();
  }
  float inv = 1.0f / red[0];
  dmout[(size_t)row * NL + threadIdx.x]       = (bf16)(e0 * inv);
  dmout[(size_t)row * NL + threadIdx.x + 256] = (bf16)(e1 * inv);
}

// ---------------------------------------------------------------- generic bf16 WMMA GEMM
// C = scale * A(MxK) @ Bn(NxK)^T  [+bias][relu][+resid]; per-wave 16x64 strip.
__global__ __launch_bounds__(256) void gemm_bf16_kernel(
    const bf16* __restrict__ A, size_t sA, int lda,
    const bf16* __restrict__ Bn, size_t sB, int ldb,
    bf16* __restrict__ obf, size_t sObf, int ldo, int transO,
    float* __restrict__ of, size_t sOf,
    const float* __restrict__ resid, size_t sR,
    const float* __restrict__ bias, int relu,
    int M, int N, int K, float scale)
{
  int batch = blockIdx.y;
  int tile  = blockIdx.x * 8 + (threadIdx.x >> 5);   // tiles over (M/16) x (N/64)
  int ntn = N / 64;
  int mt = tile / ntn, nt4 = tile % ntn;
  if (mt >= M / 16) return;
  int lane = threadIdx.x & 31, khalf = lane >> 4, ln = lane & 15;

  const bf16* Arow = A  + (size_t)batch * sA + (size_t)(mt * 16 + ln) * lda;
  const bf16* Brow = Bn + (size_t)batch * sB + (size_t)(nt4 * 64 + ln) * ldb;

  v8f acc[4];
  for (int nn = 0; nn < 4; ++nn) acc[nn] = zero8();
#pragma unroll 1
  for (int kk = 0; kk < K; kk += 32) {
    if (kk + 128 < K) {
      __builtin_prefetch(Arow + kk + 128, 0, 1);
      __builtin_prefetch(Brow + kk + 128, 0, 1);
    }
    v16bf af = ldfrag(Arow + kk, khalf);
#pragma unroll
    for (int nn = 0; nn < 4; ++nn) {
      v16bf bf = ldfrag(Brow + (size_t)nn * 16 * ldb + kk, khalf);
      acc[nn] = wmma_bf16(af, bf, acc[nn]);
    }
  }
#pragma unroll
  for (int nn = 0; nn < 4; ++nn) {
#pragma unroll
    for (int r = 0; r < 8; ++r) {
      int m = mt * 16 + r + khalf * 8;
      int n = nt4 * 64 + nn * 16 + ln;
      float v = acc[nn][r] * scale;
      if (bias)  v += bias[n];
      if (relu)  v = fmaxf(v, 0.0f);
      if (resid) v += resid[(size_t)batch * sR + (size_t)m * N + n];
      if (of)    of[(size_t)batch * sOf + (size_t)m * N + n] = v;
      if (obf) {
        size_t i = transO ? ((size_t)n * ldo + m) : ((size_t)m * ldo + n);
        obf[(size_t)batch * sObf + i] = (bf16)v;
      }
    }
  }
}

// ---------------------------------------------------------------- attention: q@k^T + q2@k2^T, masked tanh
// per-wave 16x32 strip (2-way N block), dual accumulation.
__global__ __launch_bounds__(256) void attn_kernel(
    const bf16* __restrict__ q, const bf16* __restrict__ k,
    const bf16* __restrict__ q2, const bf16* __restrict__ k2,
    const int* __restrict__ lens,
    float* __restrict__ attn_f, bf16* __restrict__ attn_bf)
{
  int b = blockIdx.y;
  int tile = blockIdx.x * 8 + (threadIdx.x >> 5);  // tiles over (NL/16) x (NL/32)
  int mt = tile >> 4, nt2 = tile & 15;
  int lane = threadIdx.x & 31, khalf = lane >> 4, ln = lane & 15;
  size_t sb = (size_t)b * NL * ND;
  const bf16* qa  = q  + sb + (size_t)(mt * 16 + ln) * ND;
  const bf16* q2a = q2 + sb + (size_t)(mt * 16 + ln) * ND;
  const bf16* kb  = k  + sb + (size_t)(nt2 * 32 + ln) * ND;
  const bf16* k2b = k2 + sb + (size_t)(nt2 * 32 + ln) * ND;

  v8f acc[2];
  acc[0] = zero8(); acc[1] = zero8();
#pragma unroll 1
  for (int kk = 0; kk < ND; kk += 32) {
    v16bf a0 = ldfrag(qa + kk, khalf);
    acc[0] = wmma_bf16(a0, ldfrag(kb + kk, khalf), acc[0]);
    acc[1] = wmma_bf16(a0, ldfrag(kb + (size_t)16 * ND + kk, khalf), acc[1]);
    v16bf a1 = ldfrag(q2a + kk, khalf);
    acc[0] = wmma_bf16(a1, ldfrag(k2b + kk, khalf), acc[0]);
    acc[1] = wmma_bf16(a1, ldfrag(k2b + (size_t)16 * ND + kk, khalf), acc[1]);
  }
  int len = lens[b];
#pragma unroll
  for (int c = 0; c < 2; ++c) {
#pragma unroll
    for (int r = 0; r < 8; ++r) {
      int l = mt * 16 + r + khalf * 8;
      int m = nt2 * 32 + c * 16 + ln;
      float v = (l < len && m < len) ? tanhf(acc[c][r]) : 0.0f;
      size_t i = (size_t)b * NL * NL + (size_t)l * NL + m;
      attn_f[i]  = v;
      attn_bf[i] = (bf16)v;
    }
  }
}

// ---------------------------------------------------------------- LayerNorm (in place on f32, bf16 copy)
__global__ __launch_bounds__(256) void ln_kernel(
    float* __restrict__ att, const float* __restrict__ g, const float* __restrict__ be,
    bf16* __restrict__ outbf)
{
  int row = blockIdx.x;
  __shared__ float red[256];
  size_t base = (size_t)row * ND;
  float v0 = att[base + threadIdx.x];
  float v1 = att[base + threadIdx.x + 256];
  red[threadIdx.x] = v0 + v1; __syncthreads();
  for (int s = 128; s > 0; s >>= 1) {
    if (threadIdx.x < s) red[threadIdx.x] += red[threadIdx.x + s];
    __syncthreads();
  }
  float mean = red[0] * (1.0f / ND); __syncthreads();
  float d0 = v0 - mean, d1 = v1 - mean;
  red[threadIdx.x] = d0 * d0 + d1 * d1; __syncthreads();
  for (int s = 128; s > 0; s >>= 1) {
    if (threadIdx.x < s) red[threadIdx.x] += red[threadIdx.x + s];
    __syncthreads();
  }
  float inv = rsqrtf(red[0] * (1.0f / ND) + 1e-6f);
  float o0 = d0 * inv * g[threadIdx.x]       + be[threadIdx.x];
  float o1 = d1 * inv * g[threadIdx.x + 256] + be[threadIdx.x + 256];
  att[base + threadIdx.x]        = o0;
  att[base + threadIdx.x + 256]  = o1;
  outbf[base + threadIdx.x]       = (bf16)o0;
  outbf[base + threadIdx.x + 256] = (bf16)o1;
}

// ---------------------------------------------------------------- final FC (N=1)
__global__ __launch_bounds__(256) void fc_kernel(
    const bf16* __restrict__ h1, const float* __restrict__ W2, const float* __restrict__ b2,
    float* __restrict__ fc, int Mtot)
{
  int row  = blockIdx.x * 8 + (threadIdx.x >> 5);
  int lane = threadIdx.x & 31;
  if (row >= Mtot) return;
  float s = 0.0f;
  for (int h = lane; h < NH; h += 32) s += (float)h1[(size_t)row * NH + h] * W2[h];
  for (int o = 16; o > 0; o >>= 1) s += __shfl_down(s, o, 32);
  if (lane == 0) fc[row] = s + b2[0];
}

// ================================================================ host
extern "C" void kernel_launch(void* const* d_in, const int* in_sizes, int n_in,
                              void* d_out, int out_size, void* d_ws, size_t ws_size,
                              hipStream_t stream)
{
  const float* x      = (const float*)d_in[0];
  const int*   lens   = (const int*)  d_in[1];
  const float* timed  = (const float*)d_in[2];
  const float* pa     = (const float*)d_in[3];
  const float* pbb    = (const float*)d_in[6];
  const float* Wih_f  = (const float*)d_in[7];
  const float* Whh_f  = (const float*)d_in[8];
  const float* bias_f = (const float*)d_in[9];
  const float* Wih_b  = (const float*)d_in[10];
  const float* Whh_b  = (const float*)d_in[11];
  const float* bias_b = (const float*)d_in[12];
  const float* wq     = (const float*)d_in[13];
  const float* wk     = (const float*)d_in[14];
  const float* wv     = (const float*)d_in[15];
  const float* wq2    = (const float*)d_in[16];
  const float* wk2    = (const float*)d_in[17];
  const float* wfc    = (const float*)d_in[18];
  const float* ln_g   = (const float*)d_in[19];
  const float* ln_b   = (const float*)d_in[20];
  const float* W1     = (const float*)d_in[21];
  const float* b1     = (const float*)d_in[22];
  const float* W2     = (const float*)d_in[23];
  const float* b2     = (const float*)d_in[24];

  float* fc_out   = (float*)d_out;
  float* att_out  = fc_out + (size_t)NB * NL;
  float* attn_out = att_out + (size_t)NB * NL * ND;

  char* wsb = (char*)d_ws;
  size_t off = 0;
  auto alloc = [&](size_t bytes) -> char* {
    off = (off + 255) & ~(size_t)255;
    char* p = wsb + off; off += bytes; return p;
  };
  const size_t BLD = (size_t)NB * NL * ND;
  const size_t BLL = (size_t)NB * NL * NL;

  float* lstm_f   = (float*)alloc(BLD * 4);
  bf16*  lstm_bf  = (bf16*) alloc(BLD * 2);
  bf16*  p_q      = (bf16*) alloc(BLD * 2);
  bf16*  p_k      = (bf16*) alloc(BLD * 2);
  bf16*  p_vT     = (bf16*) alloc(BLD * 2);
  bf16*  p_q2inT  = (bf16*) alloc(BLD * 2);
  bf16*  p_k2inT  = (bf16*) alloc(BLD * 2);
  bf16*  dm_bf    = (bf16*) alloc(BLL * 2);
  bf16*  attn_bf  = (bf16*) alloc(BLL * 2);
  bf16*  x_bf     = (bf16*) alloc((size_t)NB * NL * NIN * 2);
  bf16*  wihf_bf  = (bf16*) alloc((size_t)4 * NH * NIN * 2);
  bf16*  whhf_bf  = (bf16*) alloc((size_t)4 * NH * NH * 2);
  bf16*  wihb_bf  = (bf16*) alloc((size_t)4 * NH * NIN * 2);
  bf16*  whhb_bf  = (bf16*) alloc((size_t)4 * NH * NH * 2);
  bf16*  wq_bf    = (bf16*) alloc((size_t)ND * ND * 2);
  bf16*  wk_bf    = (bf16*) alloc((size_t)ND * ND * 2);
  bf16*  wv_bf    = (bf16*) alloc((size_t)ND * ND * 2);
  bf16*  wq2_bf   = (bf16*) alloc((size_t)ND * ND * 2);
  bf16*  wk2_bf   = (bf16*) alloc((size_t)ND * ND * 2);
  bf16*  wfc_bf   = (bf16*) alloc((size_t)ND * ND * 2);
  bf16*  w1_bf    = (bf16*) alloc((size_t)NH * ND * 2);

  // buffer reuse (sequential lifetimes)
  bf16* p_q2       = lstm_bf;   // lstm_bf dead after projections
  bf16* p_k2       = p_q2inT;   // q2inT dead after GEMM 6
  bf16* p_out1     = p_q;       // q dead after attn
  bf16* p_h1       = p_k;       // k dead after attn
  bf16* p_attoutbf = p_k2inT;   // k2inT (and alias k2) dead before LN

  auto cvt = [&](const float* s, bf16* d, size_t n) {
    cvt_bf16_kernel<<<dim3((unsigned)((n + 255) / 256)), 256, 0, stream>>>(s, d, (int)n);
  };
  cvt(x,     x_bf,    (size_t)NB * NL * NIN);
  cvt(Wih_f, wihf_bf, (size_t)4 * NH * NIN);
  cvt(Whh_f, whhf_bf, (size_t)4 * NH * NH);
  cvt(Wih_b, wihb_bf, (size_t)4 * NH * NIN);
  cvt(Whh_b, whhb_bf, (size_t)4 * NH * NH);
  cvt(wq,  wq_bf,  (size_t)ND * ND);
  cvt(wk,  wk_bf,  (size_t)ND * ND);
  cvt(wv,  wv_bf,  (size_t)ND * ND);
  cvt(wq2, wq2_bf, (size_t)ND * ND);
  cvt(wk2, wk2_bf, (size_t)ND * ND);
  cvt(wfc, wfc_bf, (size_t)ND * ND);
  cvt(W1,  w1_bf,  (size_t)NH * ND);

  (void)hipMemsetAsync(lstm_f, 0, BLD * 4, stream);
  (void)hipMemsetAsync(lstm_bf, 0, BLD * 2, stream);

  lstm_kernel<<<4, 512, 0, stream>>>(x_bf, lens, bias_f, bias_b,
                                     wihf_bf, whhf_bf, wihb_bf, whhb_bf,
                                     lstm_f, lstm_bf);

  dm_kernel<<<NB * NL, 256, 0, stream>>>(timed, lens, pa, pbb, dm_bf);

  const float invtemp = 1.0f / sqrtf((float)ND);
  const int Mfull = NB * NL;

  auto gemm = [&](const bf16* A, size_t sA, int lda, const bf16* Bn, size_t sB, int ldb,
                  bf16* obf, size_t sObf, int ldo, int transO,
                  float* of, size_t sOf, const float* resid, size_t sR,
                  const float* bias, int relu, int M, int N, int K, float scale, int nb) {
    int tiles = (M / 16) * (N / 64);
    gemm_bf16_kernel<<<dim3(tiles / 8, nb), 256, 0, stream>>>(
        A, sA, lda, Bn, sB, ldb, obf, sObf, ldo, transO, of, sOf, resid, sR, bias, relu, M, N, K, scale);
  };

  // 1,2: q (scaled), k
  gemm(lstm_bf, 0, ND, wq_bf, 0, ND, p_q, 0, ND, 0, nullptr, 0, nullptr, 0, nullptr, 0, Mfull, ND, ND, invtemp, 1);
  gemm(lstm_bf, 0, ND, wk_bf, 0, ND, p_k, 0, ND, 0, nullptr, 0, nullptr, 0, nullptr, 0, Mfull, ND, ND, 1.0f, 1);
  // 3,4,5: vT, q2inT, k2inT (batched, transposed store)
  gemm(lstm_bf, (size_t)NL * ND, ND, wv_bf,  0, ND, p_vT,    (size_t)ND * NL, NL, 1, nullptr, 0, nullptr, 0, nullptr, 0, NL, ND, ND, 1.0f, NB);
  gemm(lstm_bf, (size_t)NL * ND, ND, wq2_bf, 0, ND, p_q2inT, (size_t)ND * NL, NL, 1, nullptr, 0, nullptr, 0, nullptr, 0, NL, ND, ND, 1.0f, NB);
  gemm(lstm_bf, (size_t)NL * ND, ND, wk2_bf, 0, ND, p_k2inT, (size_t)ND * NL, NL, 1, nullptr, 0, nullptr, 0, nullptr, 0, NL, ND, ND, 1.0f, NB);
  // 6,7: q2 = dm@q2in (scaled), k2 = dm@k2in
  gemm(dm_bf, (size_t)NL * NL, NL, p_q2inT, (size_t)ND * NL, NL, p_q2, (size_t)NL * ND, ND, 0, nullptr, 0, nullptr, 0, nullptr, 0, NL, ND, NL, invtemp, NB);
  gemm(dm_bf, (size_t)NL * NL, NL, p_k2inT, (size_t)ND * NL, NL, p_k2, (size_t)NL * ND, ND, 0, nullptr, 0, nullptr, 0, nullptr, 0, NL, ND, NL, 1.0f, NB);
  // 8: attn = tanh(masked(q@k^T + q2@k2^T))
  attn_kernel<<<dim3((NL / 16) * (NL / 32) / 8, NB), 256, 0, stream>>>(p_q, p_k, p_q2, p_k2, lens, attn_out, attn_bf);
  // 9: out1 = attn @ v
  gemm(attn_bf, (size_t)NL * NL, NL, p_vT, (size_t)ND * NL, NL, p_out1, (size_t)NL * ND, ND, 0, nullptr, 0, nullptr, 0, nullptr, 0, NL, ND, NL, 1.0f, NB);
  // 10: att_pre = out1 @ wfc^T + lstm_out (f32, straight into d_out att region)
  gemm(p_out1, 0, ND, wfc_bf, 0, ND, nullptr, 0, 0, 0, att_out, 0, lstm_f, 0, nullptr, 0, Mfull, ND, ND, 1.0f, 1);
  // 11: LayerNorm in-place + bf16 copy
  ln_kernel<<<Mfull, 256, 0, stream>>>(att_out, ln_g, ln_b, p_attoutbf);
  // 12: h1 = relu(att_out @ W1^T + b1)
  gemm(p_attoutbf, 0, ND, w1_bf, 0, ND, p_h1, 0, NH, 0, nullptr, 0, nullptr, 0, b1, 1, Mfull, NH, ND, 1.0f, 1);
  // 13: fc = h1 @ W2^T + b2
  fc_kernel<<<Mfull / 8, 256, 0, stream>>>(p_h1, W2, b2, fc_out, Mfull);
}
